// LayerNormGRU_33887291966208
// MI455X (gfx1250) — matrix operand order repaired and checked by reading, not compiled, not run on hardware
//
#include <hip/hip_runtime.h>

#define Hd   1024
#define Bsz  64
#define Tsz  512
#define Lsz  2
#define EPSV 1e-5f

typedef __attribute__((ext_vector_type(16))) __bf16         v16bf;
typedef __attribute__((ext_vector_type(8)))  float          v8f;
typedef __attribute__((ext_vector_type(4)))  unsigned int   v4u;
typedef __attribute__((ext_vector_type(8)))  unsigned int   v8u;
typedef __attribute__((ext_vector_type(4)))  unsigned short v4us;

// ---------------- helpers ----------------

__device__ __forceinline__ unsigned short f2bf(float f) {
  unsigned int u = __builtin_bit_cast(unsigned int, f);
  u += 0x7FFFu + ((u >> 16) & 1u);           // round-to-nearest-even
  return (unsigned short)(u >> 16);
}

// A-operand fragment (16x32 bf16, M x K) for v_wmma_f32_16x16x32_bf16.
// Lanes 0-15:  row M=m0+lane,    K = k0+0..15
// Lanes 16-31: row M=m0+lane-16, K = k0+16..31
// Activations stored bf16 with row stride Hd -> two contiguous b128 loads.
__device__ __forceinline__ v16bf load_a_bf(const unsigned short* __restrict__ base,
                                           int m0, int k0) {
  const int lane = threadIdx.x & 31;
  const int m    = m0 + (lane & 15);
  const int kb   = (lane & 16) ? 16 : 0;
  const unsigned short* p = base + ((long long)m << 10) + (k0 + kb);
  const v4u lo = *(const v4u*)(p);        // 8 bf16: K = k0+kb .. +7
  const v4u hi = *(const v4u*)(p + 8);    // 8 bf16: K = k0+kb+8 .. +15
  const v8u w  = __builtin_shufflevector(lo, hi, 0, 1, 2, 3, 4, 5, 6, 7);
  return __builtin_bit_cast(v16bf, w);
}

// B-operand fragment (32x16 bf16, K x N). Weights stored transposed [out][in]
// so each lane's 16 consecutive K values at fixed N are one 32-byte load.
// Lanes 0-15:  N=n0+lane,    K = k0+0..15
// Lanes 16-31: N=n0+lane-16, K = k0+16..31
__device__ __forceinline__ v16bf load_b_bf(const unsigned short* __restrict__ Wt,
                                           int n0, int k0) {
  const int lane = threadIdx.x & 31;
  const unsigned short* q =
      Wt + ((long long)(n0 + (lane & 15)) << 10) + (k0 + ((lane & 16) ? 16 : 0));
  return *(const v16bf*)q;
}

__device__ __forceinline__ v8f wmma_bf16(v16bf a, v16bf b, v8f c) {
  return __builtin_amdgcn_wmma_f32_16x16x32_bf16(false, a, false, b,
                                                 (short)0, c, false, false);
}

// ---------------- prep kernels ----------------

// Convert Wi/Wh fp32 [l][g][in][out] -> bf16 transposed [l][g][out][in]
__global__ __launch_bounds__(256) void wprep_kernel(
    const float* __restrict__ Wi, const float* __restrict__ Wh,
    unsigned short* __restrict__ WiT, unsigned short* __restrict__ WhT) {
  const long long id = (long long)blockIdx.x * 256 + threadIdx.x;
  const long long per = (long long)Lsz * 3 * Hd * Hd;            // 6291456
  if (id >= 2 * per) return;
  const int sel = id >= per;
  const long long r   = id - (long long)sel * per;
  const long long mat = r >> 20;                                 // (l*3+g)
  const long long rem = r & 1048575;
  const long long i   = rem >> 10;
  const long long o   = rem & 1023;
  const float*    src = sel ? Wh  : Wi;
  unsigned short* dst = sel ? WhT : WiT;
  dst[(mat << 20) + (o << 10) + i] = f2bf(src[(mat << 20) + (i << 10) + o]);
}

// init h (fp32 + bf16) and bf16 x row for t=0
__global__ __launch_bounds__(256) void hinit_kernel(
    const float* __restrict__ h0, const float* __restrict__ x,
    float* __restrict__ hstate, unsigned short* __restrict__ hbf,
    unsigned short* __restrict__ xbf0) {
  const int idx = blockIdx.x * 256 + threadIdx.x;
  if (idx < Lsz * Bsz * Hd) {
    const float v = h0[idx];
    hstate[idx] = v;
    hbf[idx]    = f2bf(v);
  }
  if (idx < Bsz * Hd) {
    const int b = idx >> 10, j = idx & 1023;
    xbf0[idx] = f2bf(x[(long long)b * Tsz * Hd + j]);   // x[b, 0, j]
  }
}

__global__ __launch_bounds__(256) void finalize_kernel(const float* __restrict__ hstate,
                                                       float* __restrict__ out) {
  const int idx = blockIdx.x * 256 + threadIdx.x;
  if (idx < Lsz * Bsz * Hd) out[(long long)Bsz * Tsz * Hd + idx] = hstate[idx];
}

// ---------------- K1: per-step gate GEMMs ----------------
// Virtual N = 4096: vg 0 (r) and 1 (z) accumulate gi+gh into pre;
// vg 2 writes gi_n to pre; vg 3 writes gh_n to ghn.
__global__ __launch_bounds__(256) void k1_gemm(
    const unsigned short* __restrict__ Ai,   // bf16 input rows, stride Hd
    const unsigned short* __restrict__ Ah,   // bf16 recurrent rows, stride Hd
    const unsigned short* __restrict__ WiT,  // bf16 [3][out][in]
    const unsigned short* __restrict__ WhT,
    float* __restrict__ pre,                 // [B][3][Hd]
    float* __restrict__ ghn) {               // [B][Hd]
  const int wave = threadIdx.x >> 5;
  const int tile = blockIdx.x * 8 + wave;                // 0..1023
  const int m0 = (tile & 3) * 16;
  const int n0 = (tile >> 2) * 16;                       // 0..4095
  const int vg = n0 >> 10;
  const int o0 = n0 & 1023;

  v8f acc0 = {}, acc1 = {};                              // dual accumulators
  if (vg <= 2) {                                         // input projection
    const unsigned short* Wt = WiT + ((long long)vg << 20);
#pragma unroll 2
    for (int k0 = 0; k0 < Hd; k0 += 64) {
      acc0 = wmma_bf16(load_a_bf(Ai, m0, k0),      load_b_bf(Wt, o0, k0),      acc0);
      acc1 = wmma_bf16(load_a_bf(Ai, m0, k0 + 32), load_b_bf(Wt, o0, k0 + 32), acc1);
    }
  }
  if (vg != 2) {                                         // recurrent projection
    const int g = (vg == 3) ? 2 : vg;
    const unsigned short* Wt = WhT + ((long long)g << 20);
#pragma unroll 2
    for (int k0 = 0; k0 < Hd; k0 += 64) {
      acc0 = wmma_bf16(load_a_bf(Ah, m0, k0),      load_b_bf(Wt, o0, k0),      acc0);
      acc1 = wmma_bf16(load_a_bf(Ah, m0, k0 + 32), load_b_bf(Wt, o0, k0 + 32), acc1);
    }
  }
  acc0 += acc1;

  // C/D layout: VGPR r -> (M = r + 8*(lane>=16), N = lane&15)
  const int lane = threadIdx.x & 31;
  const int n  = o0 + (lane & 15);
  const int mb = m0 + ((lane & 16) ? 8 : 0);
  if (vg == 3) {
#pragma unroll
    for (int r = 0; r < 8; ++r) ghn[(long long)(mb + r) * Hd + n] = acc0[r];
  } else {
    float* dst = pre + (long long)vg * Hd;
#pragma unroll
    for (int r = 0; r < 8; ++r) dst[(long long)(mb + r) * (3 * Hd) + n] = acc0[r];
  }
}

// ---------------- K2: LayerNorm + gates + state update ----------------

// wave32 shuffle reduction + one cross-wave LDS pass; sh = 16 floats per call site
__device__ __forceinline__ void block_stats(float s, float s2, float* sh,
                                            float& mu, float& var) {
#pragma unroll
  for (int off = 16; off >= 1; off >>= 1) {
    s  += __shfl_xor(s,  off, 32);
    s2 += __shfl_xor(s2, off, 32);
  }
  const int wave = threadIdx.x >> 5;
  if ((threadIdx.x & 31) == 0) { sh[wave * 2] = s; sh[wave * 2 + 1] = s2; }
  __syncthreads();
  float ts = 0.f, ts2 = 0.f;
#pragma unroll
  for (int w = 0; w < 8; ++w) { ts += sh[w * 2]; ts2 += sh[w * 2 + 1]; }
  mu  = ts * (1.0f / Hd);
  var = ts2 * (1.0f / Hd) - mu * mu;
}

__global__ __launch_bounds__(256) void k2_cell(
    const float* __restrict__ pre, const float* __restrict__ ghn,
    float* __restrict__ hstate,                // [L][B][Hd] fp32
    unsigned short* __restrict__ hbf,          // [L][B][Hd] bf16
    const float* __restrict__ ln_g,            // [L][3][Hd]
    const float* __restrict__ ln_b,
    float* __restrict__ out,                   // [B][T][Hd]
    const float* __restrict__ xsrc,            // x base (for next-step convert)
    unsigned short* __restrict__ xbf_next,     // [B][Hd] or nullptr
    int l, int t) {
  __shared__ float sh[48];                     // 3 reduction scratch areas
  const int b = blockIdx.x, tid = threadIdx.x;
  const int j0 = tid * 4;

  // convert next timestep's x row to bf16 (layer 0 only)
  if (xbf_next) {
    const float4 a = *(const float4*)(xsrc + ((long long)b * Tsz + (t + 1)) * Hd + j0);
    v4us u; u[0] = f2bf(a.x); u[1] = f2bf(a.y); u[2] = f2bf(a.z); u[3] = f2bf(a.w);
    *(v4us*)(xbf_next + ((long long)b << 10) + j0) = u;
  }

  const float* pr = pre + (long long)b * (3 * Hd);
  const float* pz = pr + Hd;
  const float* pn = pr + 2 * Hd;
  const float* pg = ghn + (long long)b * Hd;
  float* hf = hstate + ((long long)l * Bsz + b) * Hd;

  float rv[4], zv[4], gi[4], gh[4], ho[4];
  {
    const float4 a = *(const float4*)(pr + j0); rv[0]=a.x; rv[1]=a.y; rv[2]=a.z; rv[3]=a.w;
    const float4 c = *(const float4*)(pz + j0); zv[0]=c.x; zv[1]=c.y; zv[2]=c.z; zv[3]=c.w;
    const float4 d = *(const float4*)(pn + j0); gi[0]=d.x; gi[1]=d.y; gi[2]=d.z; gi[3]=d.w;
    const float4 e = *(const float4*)(pg + j0); gh[0]=e.x; gh[1]=e.y; gh[2]=e.z; gh[3]=e.w;
    const float4 f = *(const float4*)(hf + j0); ho[0]=f.x; ho[1]=f.y; ho[2]=f.z; ho[3]=f.w;
  }

  float mu_r, var_r, mu_z, var_z;
  block_stats(rv[0]+rv[1]+rv[2]+rv[3],
              rv[0]*rv[0]+rv[1]*rv[1]+rv[2]*rv[2]+rv[3]*rv[3], sh, mu_r, var_r);
  block_stats(zv[0]+zv[1]+zv[2]+zv[3],
              zv[0]*zv[0]+zv[1]*zv[1]+zv[2]*zv[2]+zv[3]*zv[3], sh + 16, mu_z, var_z);
  const float iv_r = rsqrtf(var_r + EPSV);
  const float iv_z = rsqrtf(var_z + EPSV);

  const float* gR = ln_g + ((long long)l * 3) * Hd;
  const float* bR = ln_b + ((long long)l * 3) * Hd;
  const float* gZ = gR + Hd,     *bZ = bR + Hd;
  const float* gN = gR + 2 * Hd, *bN = bR + 2 * Hd;

  float ggr[4], bbr[4], ggz[4], bbz[4], ggn[4], bbn[4];
  { const float4 a = *(const float4*)(gR+j0); ggr[0]=a.x; ggr[1]=a.y; ggr[2]=a.z; ggr[3]=a.w; }
  { const float4 a = *(const float4*)(bR+j0); bbr[0]=a.x; bbr[1]=a.y; bbr[2]=a.z; bbr[3]=a.w; }
  { const float4 a = *(const float4*)(gZ+j0); ggz[0]=a.x; ggz[1]=a.y; ggz[2]=a.z; ggz[3]=a.w; }
  { const float4 a = *(const float4*)(bZ+j0); bbz[0]=a.x; bbz[1]=a.y; bbz[2]=a.z; bbz[3]=a.w; }
  { const float4 a = *(const float4*)(gN+j0); ggn[0]=a.x; ggn[1]=a.y; ggn[2]=a.z; ggn[3]=a.w; }
  { const float4 a = *(const float4*)(bN+j0); bbn[0]=a.x; bbn[1]=a.y; bbn[2]=a.z; bbn[3]=a.w; }

  float rr[4], zz[4], vv[4];
  float vs = 0.f, vs2 = 0.f;
#pragma unroll
  for (int i = 0; i < 4; ++i) {
    const float lr = (rv[i] - mu_r) * iv_r * ggr[i] + bbr[i];
    const float lz = (zv[i] - mu_z) * iv_z * ggz[i] + bbz[i];
    rr[i] = 1.0f / (1.0f + expf(-lr));
    zz[i] = 1.0f / (1.0f + expf(-lz));
    vv[i] = gi[i] + rr[i] * gh[i];
    vs += vv[i]; vs2 += vv[i] * vv[i];
  }
  float mu_n, var_n;
  block_stats(vs, vs2, sh + 32, mu_n, var_n);
  const float iv_n = rsqrtf(var_n + EPSV);

  float res[4] = {0.f, 0.f, 0.f, 0.f};
  if (l > 0) {
    const float* hprev = hstate + ((long long)(l - 1) * Bsz + b) * Hd;
    const float4 a = *(const float4*)(hprev + j0);
    res[0] = a.x; res[1] = a.y; res[2] = a.z; res[3] = a.w;
  }

  float hn[4];
#pragma unroll
  for (int i = 0; i < 4; ++i) {
    const float nn = tanhf((vv[i] - mu_n) * iv_n * ggn[i] + bbn[i]);
    hn[i] = (1.0f - zz[i]) * nn + zz[i] * ho[i] + res[i];
  }
  {
    float4 w; w.x = hn[0]; w.y = hn[1]; w.z = hn[2]; w.w = hn[3];
    *(float4*)(hf + j0) = w;
    v4us u; u[0] = f2bf(hn[0]); u[1] = f2bf(hn[1]); u[2] = f2bf(hn[2]); u[3] = f2bf(hn[3]);
    *(v4us*)(hbf + ((long long)l * Bsz + b) * Hd + j0) = u;
    if (l == Lsz - 1)
      *(float4*)(out + ((long long)b * Tsz + t) * Hd + j0) = w;
  }
}

// ---------------- launch ----------------

extern "C" void kernel_launch(void* const* d_in, const int* in_sizes, int n_in,
                              void* d_out, int out_size, void* d_ws, size_t ws_size,
                              hipStream_t stream) {
  const float* x    = (const float*)d_in[0];   // [B,T,H]
  const float* h0   = (const float*)d_in[1];   // [L,B,H]
  const float* Wi   = (const float*)d_in[2];   // [L,3,H,H]
  const float* Wh   = (const float*)d_in[3];   // [L,3,H,H]
  const float* ln_g = (const float*)d_in[4];   // [L,3,H]
  const float* ln_b = (const float*)d_in[5];   // [L,3,H]
  float* out = (float*)d_out;                  // output [B,T,H] ++ h_n [L,B,H]

  // workspace carve-up (all offsets >=256B aligned)
  char* ws = (char*)d_ws;
  unsigned short* WiT  = (unsigned short*)(ws);                 // 12,582,912 B
  unsigned short* WhT  = (unsigned short*)(ws + 12582912);      // 12,582,912 B
  float*          h_f32= (float*)         (ws + 25165824);      //    524,288 B
  unsigned short* hbf  = (unsigned short*)(ws + 25690112);      //    262,144 B
  unsigned short* xbf  = (unsigned short*)(ws + 25952256);      // 2x 131,072 B
  float*          pre  = (float*)         (ws + 26214400);      //    786,432 B
  float*          ghn  = (float*)         (ws + 27000832);      //    262,144 B

  // prep
  wprep_kernel<<<(2 * 6291456 + 255) / 256, 256, 0, stream>>>(Wi, Wh, WiT, WhT);
  hinit_kernel<<<(Lsz * Bsz * Hd + 255) / 256, 256, 0, stream>>>(h0, x, h_f32, hbf, xbf);

  const long long WMAT = (long long)3 * Hd * Hd;  // per-layer weight block (halfs)
  const long long XBUF = (long long)Bsz * Hd;     // ping-pong x buffer (halfs)

  for (int t = 0; t < Tsz; ++t) {
    unsigned short* xcur  = xbf + (t & 1) * XBUF;
    unsigned short* xnext = (t + 1 < Tsz) ? (xbf + ((t + 1) & 1) * XBUF) : nullptr;
    // layer 0: gi from x[:,t,:] (bf16), gh from h[0]
    k1_gemm<<<128, 256, 0, stream>>>(xcur, hbf, WiT, WhT, pre, ghn);
    k2_cell<<<Bsz, 256, 0, stream>>>(pre, ghn, h_f32, hbf, ln_g, ln_b, out,
                                     x, xnext, 0, t);
    // layer 1: gi from new h[0], gh from h[1]
    k1_gemm<<<128, 256, 0, stream>>>(hbf, hbf + XBUF, WiT + WMAT, WhT + WMAT, pre, ghn);
    k2_cell<<<Bsz, 256, 0, stream>>>(pre, ghn, h_f32, hbf, ln_g, ln_b, out,
                                     x, nullptr, 1, t);
  }

  finalize_kernel<<<(Lsz * Bsz * Hd + 255) / 256, 256, 0, stream>>>(h_f32, out);
}
